// ODEFunc_28561532519035
// MI455X (gfx1250) — matrix operand order repaired and checked by reading, not compiled, not run on hardware
//
#include <hip/hip_runtime.h>
#include <hip/hip_bf16.h>

typedef __attribute__((ext_vector_type(16))) _Float16 v16h;
typedef __attribute__((ext_vector_type(8)))  _Float16 v8h;
typedef __attribute__((ext_vector_type(8)))  float    v8f;
typedef __attribute__((ext_vector_type(4)))  unsigned int v4u;
typedef __attribute__((ext_vector_type(8)))  unsigned int v8u;

#define NB     4                    // samples per workgroup
#define TOKS   128                  // NB * 32 (V=25 padded to 32 per sample)
#define PA     136                  // pitch (f16) of [c][tok] layout, 272B: 16B aligned, bank-rotating
#define PB     264                  // pitch (f16) of [tok][c] layout, 528B: 16B aligned, bank-rotating
#define SMEM_A (256 * PA * 2)       // 69632 B
#define SMEM_B (TOKS * PB * 2)      // 67584 B  (also TDM staging: 51200 B per 2-sample chunk)
#define SMEM_PE (NB * 256 * 4)      // 4096 B
#define SMEM_TOTAL (SMEM_A + SMEM_B + SMEM_PE)   // 141312 B < 320 KB/WG
#define CHUNK_ELEMS (2 * 256 * 25)  // 12800 f32 per TDM chunk

static __device__ __forceinline__ v16h cat8(v8h lo, v8h hi_) {
  return __builtin_shufflevector(lo, hi_, 0,1,2,3,4,5,6,7,8,9,10,11,12,13,14,15);
}

static __device__ __forceinline__ v8f wmma_f16(v16h a, v16h b, v8f c) {
  // D = A(16x32 f16) * B(32x16 f16) + C(16x16 f32)
  return __builtin_amdgcn_wmma_f32_16x16x32_f16(false, a, false, b, (short)0, c, false, false);
}

// ---- TDM: DMA a contiguous run of nElems f32 from global into LDS (1-row 2D tile) ----
// D# per CDNA5 ISA ch.8: group0 = {flags, lds_addr, global_addr, type=2},
// group1 = {data_size=4B, tensor_dim0/1, tile_dim0/1, strides}, groups2/3 = benign 1s.
static __device__ __forceinline__ void tdm_load_x(const float* gsrc, unsigned ldsOff, int nElems) {
  unsigned long long ga = (unsigned long long)gsrc;
  v4u g0;
  g0[0] = 1u;                                             // count=1 (valid), user mode
  g0[1] = ldsOff;                                         // lds_addr (bytes)
  g0[2] = (unsigned)(ga & 0xFFFFFFFFu);                   // global_addr[31:0]
  g0[3] = (unsigned)((ga >> 32) & 0x01FFFFFFu) | (2u << 30); // global_addr[56:32] | type=2
  unsigned n = (unsigned)nElems;
  v8u g1;
  g1[0] = (2u << 16);                                     // workgroup_mask=0, data_size=2 (4B)
  g1[1] = (n & 0xFFFFu) << 16;                            // tensor_dim0[15:0] @ bits 63:48
  g1[2] = ((n >> 16) & 0xFFFFu) | (1u << 16);             // tensor_dim0[31:16] | tensor_dim1=1
  g1[3] = (n & 0xFFFFu) << 16;                            // tile_dim0 @ bits 127:112
  g1[4] = 1u;                                             // tile_dim1=1, tile_dim2=0
  g1[5] = n;                                              // tensor_dim0_stride[31:0]
  g1[6] = 0u;                                             // stride0 hi / stride1 lo
  g1[7] = 0u;
  v4u g2 = (v4u){1u, 1u, 0u, 0u};                         // tensor_dim2=1, tensor_dim3=1
  v4u g3 = (v4u){0u, 1u << 16, 0u, 0u};                   // tensor_dim4=1
  asm volatile("tensor_load_to_lds %0, %1, %2, %3"
               :: "s"(g0), "s"(g1), "s"(g2), "s"(g3) : "memory");
}

// ---- A-mix: Out_n[c,v] = sum_u H_n[c,u] * A[v,u]  (per sample, V padded to 32) ----
static __device__ __forceinline__ void amix_stage(const _Float16* __restrict__ srcA,
                                                  _Float16* __restrict__ dstB,
                                                  const v16h* bop,
                                                  int lane, int wave, int hi) {
  const int m16 = lane & 15;
  for (int j = 0; j < 16; ++j) {          // 128 tiles / 8 waves
    int f  = wave + j * 8;                // f = nl*32 + ct*2 + vt
    int vt = f & 1;
    int ct = (f >> 1) & 15;
    int nl = f >> 5;
    const _Float16* ap = srcA + (ct * 16 + m16) * PA + nl * 32 + hi * 8;
    v16h a = cat8(*(const v8h*)ap, *(const v8h*)(ap + 16));
    v8f d = {0.f, 0.f, 0.f, 0.f, 0.f, 0.f, 0.f, 0.f};
    d = wmma_f16(a, bop[vt], d);
    int v = vt * 16 + m16;
    if (v < 25) {
      _Float16* drow = dstB + (nl * 32 + v) * PB + ct * 16 + hi * 8;
      #pragma unroll
      for (int r = 0; r < 8; ++r) drow[r] = (_Float16)d[r];
    }
  }
}

// ---- 256x256 GEMM over 128 token columns, software-pipelined operand streams ----
// MODE 0: bias+lrelu -> dstLds as [c][tok] (pitch PA); MODE 1: bias+lrelu -> [tok][c] (PB);
// MODE 2: bias -> global f32.
template <int MODE>
static __device__ __forceinline__ void gemm_stage(const float* __restrict__ W,
                                                  const float* __restrict__ bias,
                                                  const _Float16* __restrict__ src,  // [tok][c] pitch PB
                                                  _Float16* __restrict__ dstLds,
                                                  float* __restrict__ outG,
                                                  int blk, int lane, int wave, int hi) {
  const int m16 = lane & 15;
  for (int oti = 0; oti < 2; ++oti) {
    const int ot = wave * 2 + oti;        // 16 o-tiles across 8 waves
    v8f acc[8];
    #pragma unroll
    for (int t = 0; t < 8; ++t) acc[t] = (v8f){0.f,0.f,0.f,0.f,0.f,0.f,0.f,0.f};

    const float* wbase = W + (size_t)(ot * 16 + m16) * 256 + hi * 8;
    float wf0[8], wf1[8];
    #pragma unroll
    for (int e = 0; e < 8; ++e) { wf0[e] = wbase[e]; wf1[e] = wbase[16 + e]; }

    for (int k = 0; k < 8; ++k) {
      v16h a;                             // A-op: K = (e&7) + (e>=8?16:0) + hi*8
      #pragma unroll
      for (int e = 0; e < 8; ++e) { a[e] = (_Float16)wf0[e]; a[e + 8] = (_Float16)wf1[e]; }
      if (k < 7) {                        // prefetch next K chunk of W while WMMAs run
        const float* wn = wbase + (k + 1) * 32;
        __builtin_prefetch(wn + 32, 0, 1);
        #pragma unroll
        for (int e = 0; e < 8; ++e) { wf0[e] = wn[e]; wf1[e] = wn[16 + e]; }
      }
      // B-op pipeline: load t+1 while WMMA t issues
      const _Float16* bp0 = src + m16 * PB + k * 32 + hi * 8;
      v16h b = cat8(*(const v8h*)bp0, *(const v8h*)(bp0 + 16));
      #pragma unroll
      for (int t = 0; t < 8; ++t) {
        v16h bn = b;
        if (t < 7) {
          const _Float16* bp = src + ((t + 1) * 16 + m16) * PB + k * 32 + hi * 8;
          bn = cat8(*(const v8h*)bp, *(const v8h*)(bp + 16));
        }
        acc[t] = wmma_f16(a, b, acc[t]);
        b = bn;
      }
    }

    float bv[8];
    const float* bb = bias + ot * 16 + hi * 8;
    #pragma unroll
    for (int r = 0; r < 8; ++r) bv[r] = bb[r];

    #pragma unroll
    for (int t = 0; t < 8; ++t) {
      int tok = t * 16 + m16;
      int u  = tok & 31;
      int nl = tok >> 5;
      #pragma unroll
      for (int r = 0; r < 8; ++r) {
        int o = ot * 16 + hi * 8 + r;     // D: M = r + hi*8
        float v = acc[t][r] + bv[r];
        if (MODE != 2) v = (v >= 0.f) ? v : 0.01f * v;   // LeakyReLU
        if (MODE == 0) {
          dstLds[o * PA + tok] = (u < 25) ? (_Float16)v : (_Float16)0.0f;
        } else if (MODE == 1) {
          dstLds[tok * PB + o] = (_Float16)v;
        } else {
          if (u < 25)
            outG[((size_t)(blk * NB + nl) * 256 + o) * 25 + u] = v;
        }
      }
    }
  }
}

__global__ __launch_bounds__(256) void ode_fused_kernel(
    const float* __restrict__ x,  const float* __restrict__ Amat,
    const float* __restrict__ w1, const float* __restrict__ b1,
    const float* __restrict__ w2, const float* __restrict__ b2,
    const float* __restrict__ wp, const float* __restrict__ bp,
    const int* __restrict__ tptr, float* __restrict__ out) {
  extern __shared__ __align__(16) char smem[];
  _Float16* bufA = (_Float16*)smem;                       // [c][tok]; later h4 as [tok][c]
  _Float16* bufB = (_Float16*)(smem + SMEM_A);            // [tok][c]; start: TDM f32 staging
  float*    peB  = (float*)(smem + SMEM_A + SMEM_B);      // [NB][256]

  const int tid  = threadIdx.x;
  const int lane = tid & 31;
  const int wave = tid >> 5;
  const int hi   = (lane >> 4) & 1;
  const int blk  = blockIdx.x;
  const int tval = tptr[0];
  const float* xblk = x + (size_t)blk * NB * 6400;

  // Kick off TDM for chunk 0 (samples 0-1) before doing any setup math.
  if (wave == 0) tdm_load_x(xblk, (unsigned)SMEM_A, CHUNK_ELEMS);

  // A^T (padded 32x32, f16) as resident B-operands: K=u=(e&7)+hi*8+(e>=8?16:0), N=v=vt*16+lane%16
  v16h bop[2];
  {
    const int vb = lane & 15;
    #pragma unroll
    for (int vt = 0; vt < 2; ++vt) {
      int v = vt * 16 + vb;
      #pragma unroll
      for (int e = 0; e < 16; ++e) {
        int u = (e & 7) + hi * 8 + (e >> 3) * 16;
        float av = (v < 25 && u < 25) ? Amat[v * 25 + u] : 0.f;
        bop[vt][e] = (_Float16)av;
      }
    }
  }

  // Sinusoidal PE rows for this block's samples: pe[row, c], row = t + (n % 64)
  {
    float dv = __expf(-(float)(tid >> 1) * 0.07195578f);  // ln(10000)/128
    #pragma unroll
    for (int nl = 0; nl < NB; ++nl) {
      int row = tval + ((blk * NB + nl) & 63);
      float ang = (float)row * dv;
      peB[nl * 256 + tid] = (tid & 1) ? __cosf(ang) : __sinf(ang);
    }
  }

  if (wave == 0) __builtin_amdgcn_s_wait_tensorcnt((short)0);
  __syncthreads();

  // Convert staged chunks: f32 x + pe -> f16 bufA [c][tok]; zero padded v=25..31
  const float* stage = (const float*)bufB;
  for (int ch = 0; ch < 2; ++ch) {
    for (int i = 0; i < 64; ++i) {
      int flat = tid + i * 256;           // [nl2:2][c:256][v:32]
      int v   = flat & 31;
      int c   = (flat >> 5) & 255;
      int nl2 = flat >> 13;
      int nl  = ch * 2 + nl2;
      float val = 0.f;
      if (v < 25)
        val = stage[(nl2 * 256 + c) * 25 + v] + peB[nl * 256 + c];
      bufA[c * PA + nl * 32 + v] = (_Float16)val;
    }
    __syncthreads();                      // staging fully consumed
    if (ch == 0) {
      if (wave == 0) {
        tdm_load_x(xblk + CHUNK_ELEMS, (unsigned)SMEM_A, CHUNK_ELEMS);
        __builtin_amdgcn_s_wait_tensorcnt((short)0);
      }
      __syncthreads();
    }
  }

  amix_stage(bufA, bufB, bop, lane, wave, hi);                       // h1 -> bufB
  __syncthreads();
  gemm_stage<0>(w1, b1, bufB, bufA, nullptr, blk, lane, wave, hi);   // h2 -> bufA [c][tok]
  __syncthreads();
  amix_stage(bufA, bufB, bop, lane, wave, hi);                       // h3 -> bufB
  __syncthreads();
  gemm_stage<1>(w2, b2, bufB, bufA, nullptr, blk, lane, wave, hi);   // h4 -> bufA [tok][c]
  __syncthreads();
  gemm_stage<2>(wp, bp, bufA, nullptr, out, blk, lane, wave, hi);    // out -> global
}

extern "C" void kernel_launch(void* const* d_in, const int* in_sizes, int n_in,
                              void* d_out, int out_size, void* d_ws, size_t ws_size,
                              hipStream_t stream) {
  const float* x  = (const float*)d_in[0];
  const float* A  = (const float*)d_in[1];
  const float* w1 = (const float*)d_in[2];
  const float* b1 = (const float*)d_in[3];
  const float* w2 = (const float*)d_in[4];
  const float* b2 = (const float*)d_in[5];
  const float* wp = (const float*)d_in[6];
  const float* bp = (const float*)d_in[7];
  const int*   t  = (const int*)d_in[8];
  float* out = (float*)d_out;
  (void)in_sizes; (void)n_in; (void)out_size; (void)d_ws; (void)ws_size;

  hipFuncSetAttribute((const void*)ode_fused_kernel,
                      hipFuncAttributeMaxDynamicSharedMemorySize, SMEM_TOTAL);
  ode_fused_kernel<<<dim3(4096 / NB), dim3(256), SMEM_TOTAL, stream>>>(
      x, A, w1, b1, w2, b2, wp, bp, t, out);
}
